// MultiHeadAttention_5025111736585
// MI455X (gfx1250) — matrix operand order repaired and checked by reading, not compiled, not run on hardware
//
#include <hip/hip_runtime.h>

typedef __bf16 bf16;
typedef __attribute__((ext_vector_type(16))) bf16  v16bf;
typedef __attribute__((ext_vector_type(8)))  float v8f;
typedef __attribute__((ext_vector_type(4)))  unsigned int v4ui;
typedef __attribute__((ext_vector_type(4)))  int v4i;
typedef __attribute__((ext_vector_type(8)))  int v8i;

#define WMMA_BF16(a, b, c) \
  __builtin_amdgcn_wmma_f32_16x16x32_bf16(false, (a), false, (b), (short)0, (c), false, false)

#if defined(__has_builtin)
#if __has_builtin(__builtin_amdgcn_tensor_load_to_lds) && \
    __has_builtin(__builtin_amdgcn_s_wait_tensorcnt)
#define HAVE_TDM 1
#endif
#endif

static constexpr int SEQ    = 2048;
static constexpr int NHEAD  = 16;
static constexpr int HDIM   = 64;
static constexpr int DIM    = 1024;
static constexpr int NBATCH = 4;
static constexpr int MROWS  = NBATCH * SEQ;   // 8192
static constexpr int KTILES = DIM / 32;       // 32

__device__ inline v8f zero8() {
  v8f z;
#pragma unroll
  for (int i = 0; i < 8; ++i) z[i] = 0.0f;
  return z;
}

// 32 contiguous bytes -> v16bf (contiguous k order: 16 consecutive k per lane, not AMD's table order)
__device__ inline v16bf ld_frag32(const bf16* p) {
  v16bf r;
  ((uint4*)&r)[0] = ((const uint4*)p)[0];
  ((uint4*)&r)[1] = ((const uint4*)p)[1];
  return r;
}

// A-fragment per 16-bit A 16x32 layout: lane elements = K{klo..klo+7, 16+klo..16+klo+7}
__device__ inline v16bf ld_afrag(const bf16* row, int klo) {
  v16bf r;
  ((uint4*)&r)[0] = *(const uint4*)(row + klo);
  ((uint4*)&r)[1] = *(const uint4*)(row + 16 + klo);
  return r;
}

#ifdef HAVE_TDM
// TDM: load a 32(key) x 64(d) bf16 tile (row stride 64 elems in memory) into LDS,
// padding 4 DWORDs after every 32 DWORDs so LDS rows land at stride 72 bf16.
__device__ inline void tdm_load_k_tile(unsigned lds_off, const bf16* gptr) {
  unsigned long long ga = (unsigned long long)(uintptr_t)gptr;
  v4ui g0;
  g0[0] = 1u;                                        // count=1, user descriptor
  g0[1] = lds_off;                                   // lds_addr (bytes)
  g0[2] = (unsigned)(ga & 0xffffffffu);              // global_addr[31:0]
  g0[3] = (unsigned)((ga >> 32) & 0x01ffffffu)       // global_addr[56:32]
          | 0x80000000u;                             // type=2 ("image")
  v8i g1;
  g1[0] = 0x00010000                                 // data_size=1 (2 bytes)
          | (1 << 20)                                // pad_enable
          | (4 << 22)                                // pad_interval: 32 DWORDs
          | (3 << 25);                               // pad_amount: 4 DWORDs
  g1[1] = (int)(64u << 16);                          // tensor_dim0 = 64 (bits 63:48)
  g1[2] = (int)(32u << 16);                          // tensor_dim1 = 32 (bits 95:80)
  g1[3] = (int)(64u << 16);                          // tile_dim0  = 64 (bits 127:112)
  g1[4] = 32;                                        // tile_dim1  = 32
  g1[5] = 64;                                        // tensor_dim0_stride = 64
  g1[6] = 0;
  g1[7] = 0;
  v4i z4 = {0, 0, 0, 0};
#if defined(__clang_major__) && __clang_major__ >= 23
  v8i z8 = {0, 0, 0, 0, 0, 0, 0, 0};
  __builtin_amdgcn_tensor_load_to_lds(g0, g1, z4, z4, z8, 0);
#else
  __builtin_amdgcn_tensor_load_to_lds(g0, g1, z4, z4, 0);
#endif
}
#endif

// ---------------------------------------------------------------------------
// QKV projection: out[b][h][s][d] = bf16( x @ W + bias ).  BM=BN=128, BK=32,
// double-buffered LDS, software-pipelined global loads.  8 waves = 2x4,
// each wave 64x32 (4x2 wmma tiles).
// ---------------------------------------------------------------------------
__global__ __launch_bounds__(256) void qkv_proj_kernel(
    const float* __restrict__ X, const float* __restrict__ W,
    const float* __restrict__ bias, bf16* __restrict__ out) {
  __shared__ bf16 Ash[2][128][40];
  __shared__ bf16 Bsh[2][128][40];   // transposed: Bsh[buf][n][k]

  const int tid = threadIdx.x;
  const int lane = tid & 31, wid = tid >> 5;
  const int lane16 = lane & 15, halfw = (lane >> 4) & 1;
  const int rw = wid >> 2, cw = wid & 3;
  const int Mbase = (int)(blockIdx.x >> 3) * 128;
  const int Nbase = (int)(blockIdx.x & 7) * 128;

  const int arow = tid >> 1, aseg = (tid & 1) * 16;
  const int bk = tid >> 3, bnseg = (tid & 7) * 16;
  const float* aptr = X + (size_t)(Mbase + arow) * DIM + aseg;
  const float* bptr = W + (size_t)bk * DIM + Nbase + bnseg;

  float4 aR[4], bR[4];
  auto load_tile = [&](int kt) {
    const float* ap = aptr + kt * 32;
    const float* bp = bptr + (size_t)kt * 32 * DIM;
#pragma unroll
    for (int i = 0; i < 4; ++i) aR[i] = *(const float4*)(ap + i * 4);
#pragma unroll
    for (int i = 0; i < 4; ++i) bR[i] = *(const float4*)(bp + i * 4);
  };
  auto stage_tile = [&](int buf) {
    bf16 tmp[16];
#pragma unroll
    for (int i = 0; i < 4; ++i) {
      tmp[4 * i + 0] = (bf16)aR[i].x; tmp[4 * i + 1] = (bf16)aR[i].y;
      tmp[4 * i + 2] = (bf16)aR[i].z; tmp[4 * i + 3] = (bf16)aR[i].w;
    }
    *(uint4*)&Ash[buf][arow][aseg]     = ((uint4*)tmp)[0];
    *(uint4*)&Ash[buf][arow][aseg + 8] = ((uint4*)tmp)[1];
#pragma unroll
    for (int i = 0; i < 4; ++i) {
      Bsh[buf][bnseg + 4 * i + 0][bk] = (bf16)bR[i].x;
      Bsh[buf][bnseg + 4 * i + 1][bk] = (bf16)bR[i].y;
      Bsh[buf][bnseg + 4 * i + 2][bk] = (bf16)bR[i].z;
      Bsh[buf][bnseg + 4 * i + 3][bk] = (bf16)bR[i].w;
    }
  };

  v8f acc[4][2];
#pragma unroll
  for (int mt = 0; mt < 4; ++mt)
#pragma unroll
    for (int nt = 0; nt < 2; ++nt) acc[mt][nt] = zero8();

  load_tile(0);
  stage_tile(0);
  __syncthreads();

  for (int kt = 0; kt < KTILES; ++kt) {
    const int cur = kt & 1;
    if (kt + 1 < KTILES) load_tile(kt + 1);

    v16bf af[4], bfr[2];
#pragma unroll
    for (int mt = 0; mt < 4; ++mt)
      af[mt] = ld_afrag(&Ash[cur][rw * 64 + mt * 16 + lane16][0], halfw * 8);
#pragma unroll
    for (int nt = 0; nt < 2; ++nt)
      bfr[nt] = ld_afrag(&Bsh[cur][cw * 32 + nt * 16 + lane16][0], halfw * 8);
#pragma unroll
    for (int mt = 0; mt < 4; ++mt)
#pragma unroll
      for (int nt = 0; nt < 2; ++nt)
        acc[mt][nt] = WMMA_BF16(af[mt], bfr[nt], acc[mt][nt]);

    if (kt + 1 < KTILES) stage_tile(cur ^ 1);
    __syncthreads();
  }

  // store: C/D layout -> [b][h][s][d] bf16
#pragma unroll
  for (int mt = 0; mt < 4; ++mt) {
#pragma unroll
    for (int nt = 0; nt < 2; ++nt) {
      const int col = Nbase + cw * 32 + nt * 16 + lane16;
      const int hh = col >> 6, dd = col & 63;
      const float bv = bias[col];
#pragma unroll
      for (int e = 0; e < 8; ++e) {
        const int row = Mbase + rw * 64 + mt * 16 + e + halfw * 8;
        const int bb = row >> 11, ss = row & (SEQ - 1);
        out[(((size_t)bb * NHEAD + hh) * SEQ + ss) * HDIM + dd] =
            (bf16)(acc[mt][nt][e] + bv);
      }
    }
  }
}

// ---------------------------------------------------------------------------
// Flash attention: grid = B*H * (S/128); 8 waves, each owns a 16-row Q tile.
// K chunk staged via Tensor Data Mover (padded LDS rows); V transposed by ALUs.
// ---------------------------------------------------------------------------
__global__ __launch_bounds__(256) void attn_kernel(
    const bf16* __restrict__ Q, const bf16* __restrict__ K,
    const bf16* __restrict__ V, const int* __restrict__ mask,
    bf16* __restrict__ ctx) {
  __shared__ bf16 Ksh[32][72];        // [key][d], stride 72 matches TDM padding
  __shared__ bf16 Vsh[64][40];        // transposed: [d][key]
  __shared__ bf16 Psh[8][16][40];     // per-wave P transpose scratch

  const int tid = threadIdx.x;
  const int lane = tid & 31, wid = tid >> 5;
  const int lane16 = lane & 15, halfw = (lane >> 4) & 1;
  const int bh = (int)(blockIdx.x >> 4);       // b*16 + h
  const int qt = (int)(blockIdx.x & 15);
  const int b = bh >> 4, h = bh & 15;
  const int s0 = qt * 128 + wid * 16;
  const float scale = 0.125f;                  // 1/sqrt(64)

  const bf16* qrow = Q + ((size_t)bh * SEQ + s0 + lane16) * HDIM;
  v16bf qf[2];
#pragma unroll
  for (int f = 0; f < 2; ++f) qf[f] = ld_afrag(qrow + f * 32, halfw * 8);

  v8f o[4];
#pragma unroll
  for (int t = 0; t < 4; ++t) o[t] = zero8();
  float mrow[8], lsum[8];
#pragma unroll
  for (int r = 0; r < 8; ++r) { mrow[r] = -1e30f; lsum[r] = 0.0f; }

  const int* maskrow = mask + (size_t)b * SEQ;
  const int key = tid >> 3, seg = (tid & 7) * 8;
#ifdef HAVE_TDM
  const unsigned ksh_off = (unsigned)(uintptr_t)(void*)&Ksh[0][0];
#endif

  for (int ci = 0; ci < SEQ / 32; ++ci) {
    const int cbase = ci * 32;
    // ---- stage V transposed (all threads)
    {
      const bf16* vrow = V + ((size_t)bh * SEQ + cbase + key) * HDIM + seg;
      uint4 vv = *(const uint4*)vrow;
      const bf16* vp = (const bf16*)&vv;
#pragma unroll
      for (int i = 0; i < 8; ++i) Vsh[seg + i][key] = vp[i];
      if (ci + 1 < SEQ / 32) __builtin_prefetch(vrow + 32 * HDIM, 0, 1);
    }
    // ---- stage K chunk (32x64 bf16) into Ksh
#ifdef HAVE_TDM
    if (wid == 0) {
      tdm_load_k_tile(ksh_off, K + ((size_t)bh * SEQ + cbase) * HDIM);
      __builtin_amdgcn_s_wait_tensorcnt(0);
    }
#else
    {
      const bf16* krow = K + ((size_t)bh * SEQ + cbase + key) * HDIM + seg;
      *(uint4*)&Ksh[key][seg] = *(const uint4*)krow;
    }
#endif
    __syncthreads();

    // ---- S = scale * Q K^T for 32 keys
    v8f sc[2];
#pragma unroll
    for (int u = 0; u < 2; ++u) {
      sc[u] = zero8();
#pragma unroll
      for (int f = 0; f < 2; ++f) {
        v16bf kb = ld_afrag(&Ksh[u * 16 + lane16][f * 32], halfw * 8);
        sc[u] = WMMA_BF16(qf[f], kb, sc[u]);
      }
    }

    const int mk0 = maskrow[cbase + lane16];
    const int mk1 = maskrow[cbase + 16 + lane16];
    float s0a[8], s1a[8];
#pragma unroll
    for (int r = 0; r < 8; ++r) {
      s0a[r] = mk0 ? sc[0][r] * scale : -1e9f;
      s1a[r] = mk1 ? sc[1][r] * scale : -1e9f;
    }

    // ---- online softmax (rows of a half-wave live in lanes 0-15 / 16-31)
#pragma unroll
    for (int r = 0; r < 8; ++r) {
      float mx = fmaxf(s0a[r], s1a[r]);
#pragma unroll
      for (int off = 8; off >= 1; off >>= 1) mx = fmaxf(mx, __shfl_xor(mx, off, 16));
      const float mn = fmaxf(mrow[r], mx);
      const float corr = __expf(mrow[r] - mn);
      const float p0 = __expf(s0a[r] - mn);
      const float p1 = __expf(s1a[r] - mn);
      float sm = p0 + p1;
#pragma unroll
      for (int off = 8; off >= 1; off >>= 1) sm += __shfl_xor(sm, off, 16);
      lsum[r] = lsum[r] * corr + sm;
      mrow[r] = mn;
      const int m = r + halfw * 8;
      Psh[wid][m][lane16]      = (bf16)p0;
      Psh[wid][m][16 + lane16] = (bf16)p1;
#pragma unroll
      for (int t = 0; t < 4; ++t) o[t][r] = o[t][r] * corr;
    }
    asm volatile("s_wait_dscnt 0" ::: "memory");   // per-wave P transpose visible

    // ---- O += P (16x32) @ V (32x64)
    v16bf pf = ld_afrag(&Psh[wid][lane16][0], halfw * 8);
#pragma unroll
    for (int t = 0; t < 4; ++t) {
      v16bf vb = ld_afrag(&Vsh[t * 16 + lane16][0], halfw * 8);
      o[t] = WMMA_BF16(pf, vb, o[t]);
    }
    __syncthreads();   // before next chunk overwrites Ksh/Vsh
  }

  // ---- normalize and store ctx[b*S+s][h*64+d]
#pragma unroll
  for (int t = 0; t < 4; ++t) {
    const int col = h * HDIM + t * 16 + lane16;
#pragma unroll
    for (int r = 0; r < 8; ++r) {
      const int ss = s0 + r + halfw * 8;
      const float inv = 1.0f / fmaxf(lsum[r], 1e-20f);
      ctx[((size_t)b * SEQ + ss) * DIM + col] = (bf16)(o[t][r] * inv);
    }
  }
}

// ---------------------------------------------------------------------------
// Output projection: out(fp32) = ctx_bf16 @ Wo + bo.  Same pipelined GEMM.
// ---------------------------------------------------------------------------
__global__ __launch_bounds__(256) void out_proj_kernel(
    const bf16* __restrict__ A, const float* __restrict__ W,
    const float* __restrict__ bias, float* __restrict__ out) {
  __shared__ bf16 Ash[2][128][40];
  __shared__ bf16 Bsh[2][128][40];

  const int tid = threadIdx.x;
  const int lane = tid & 31, wid = tid >> 5;
  const int lane16 = lane & 15, halfw = (lane >> 4) & 1;
  const int rw = wid >> 2, cw = wid & 3;
  const int Mbase = (int)(blockIdx.x >> 3) * 128;
  const int Nbase = (int)(blockIdx.x & 7) * 128;

  const int arow = tid >> 1, aseg = (tid & 1) * 16;
  const int bk = tid >> 3, bnseg = (tid & 7) * 16;
  const bf16* aptr = A + (size_t)(Mbase + arow) * DIM + aseg;
  const float* bptr = W + (size_t)bk * DIM + Nbase + bnseg;

  uint4 aR[2];
  float4 bR[4];
  auto load_tile = [&](int kt) {
    const bf16* ap = aptr + kt * 32;
    const float* bp = bptr + (size_t)kt * 32 * DIM;
    aR[0] = ((const uint4*)ap)[0];
    aR[1] = ((const uint4*)ap)[1];
#pragma unroll
    for (int i = 0; i < 4; ++i) bR[i] = *(const float4*)(bp + i * 4);
  };
  auto stage_tile = [&](int buf) {
    *(uint4*)&Ash[buf][arow][aseg]     = aR[0];
    *(uint4*)&Ash[buf][arow][aseg + 8] = aR[1];
#pragma unroll
    for (int i = 0; i < 4; ++i) {
      Bsh[buf][bnseg + 4 * i + 0][bk] = (bf16)bR[i].x;
      Bsh[buf][bnseg + 4 * i + 1][bk] = (bf16)bR[i].y;
      Bsh[buf][bnseg + 4 * i + 2][bk] = (bf16)bR[i].z;
      Bsh[buf][bnseg + 4 * i + 3][bk] = (bf16)bR[i].w;
    }
  };

  v8f acc[4][2];
#pragma unroll
  for (int mt = 0; mt < 4; ++mt)
#pragma unroll
    for (int nt = 0; nt < 2; ++nt) acc[mt][nt] = zero8();

  load_tile(0);
  stage_tile(0);
  __syncthreads();

  for (int kt = 0; kt < KTILES; ++kt) {
    const int cur = kt & 1;
    if (kt + 1 < KTILES) load_tile(kt + 1);

    v16bf af[4], bfr[2];
#pragma unroll
    for (int mt = 0; mt < 4; ++mt)
      af[mt] = ld_afrag(&Ash[cur][rw * 64 + mt * 16 + lane16][0], halfw * 8);
#pragma unroll
    for (int nt = 0; nt < 2; ++nt)
      bfr[nt] = ld_afrag(&Bsh[cur][cw * 32 + nt * 16 + lane16][0], halfw * 8);
#pragma unroll
    for (int mt = 0; mt < 4; ++mt)
#pragma unroll
      for (int nt = 0; nt < 2; ++nt)
        acc[mt][nt] = WMMA_BF16(af[mt], bfr[nt], acc[mt][nt]);

    if (kt + 1 < KTILES) stage_tile(cur ^ 1);
    __syncthreads();
  }

#pragma unroll
  for (int mt = 0; mt < 4; ++mt) {
#pragma unroll
    for (int nt = 0; nt < 2; ++nt) {
      const int col = Nbase + cw * 32 + nt * 16 + lane16;
      const float bv = bias[col];
#pragma unroll
      for (int e = 0; e < 8; ++e) {
        const int row = Mbase + rw * 64 + mt * 16 + e + halfw * 8;
        out[(size_t)row * DIM + col] = acc[mt][nt][e] + bv;
      }
    }
  }
}

// ---------------------------------------------------------------------------
extern "C" void kernel_launch(void* const* d_in, const int* in_sizes, int n_in,
                              void* d_out, int out_size, void* d_ws, size_t ws_size,
                              hipStream_t stream) {
  const float* x   = (const float*)d_in[0];
  const int*  mask = (const int*)  d_in[1];
  const float* Wq  = (const float*)d_in[2];
  const float* bq  = (const float*)d_in[3];
  const float* Wk  = (const float*)d_in[4];
  const float* bk  = (const float*)d_in[5];
  const float* Wv  = (const float*)d_in[6];
  const float* bv  = (const float*)d_in[7];
  const float* Wo  = (const float*)d_in[8];
  const float* bo  = (const float*)d_in[9];
  float* out = (float*)d_out;

  const size_t elems = (size_t)MROWS * DIM;  // 8192*1024
  bf16* qb  = (bf16*)d_ws;
  bf16* kb  = qb + elems;
  bf16* vb  = kb + elems;
  bf16* ctx = vb + elems;

  const dim3 gemm_grid(512), blk(256);
  qkv_proj_kernel<<<gemm_grid, blk, 0, stream>>>(x, Wq, bq, qb);
  qkv_proj_kernel<<<gemm_grid, blk, 0, stream>>>(x, Wk, bk, kb);
  qkv_proj_kernel<<<gemm_grid, blk, 0, stream>>>(x, Wv, bv, vb);
  attn_kernel<<<dim3(NBATCH * NHEAD * (SEQ / 128)), blk, 0, stream>>>(qb, kb, vb, mask, ctx);
  out_proj_kernel<<<gemm_grid, blk, 0, stream>>>(ctx, Wo, bo, out);
}
